// CustomMultiHeadSelfAttention_48790828482608
// MI455X (gfx1250) — compile-verified, matrix operand order checked by reading, not run on hardware
//
#include <hip/hip_runtime.h>

#define B_ 2
#define N_ 2048
#define C_ 1024
#define H_ 16
#define D_ 64

typedef __attribute__((ext_vector_type(16))) __bf16        v16bf;
typedef __attribute__((ext_vector_type(8)))  float         v8f;
typedef __attribute__((ext_vector_type(4)))  unsigned int  u32x4;

union Frag { v16bf v; u32x4 q[2]; };

__device__ __forceinline__ unsigned short f2bf(float f) {
  unsigned int u = __float_as_uint(f);
  u += 0x7FFFu + ((u >> 16) & 1u);          // round-to-nearest-even
  return (unsigned short)(u >> 16);
}

// Async DMA: 16 bytes global -> LDS per active lane (ASYNCcnt-tracked, no VGPR staging).
__device__ __forceinline__ void async_copy16(unsigned lds_off, const void* g) {
  asm volatile("global_load_async_to_lds_b128 %0, %1, off"
               :: "v"(lds_off), "v"((unsigned long long)g)
               : "memory");
}
__device__ __forceinline__ void wait_async_all() {
  asm volatile("s_wait_asynccnt 0" ::: "memory");
}

// Load one 16x32 bf16 A/B fragment from a row-major [rows][ld] tile (LDS or global).
// ISA 16-bit layout: lanes 0-15 take k = k0+hi*8+{0..7} and k0+16+hi*8+{0..7}.
__device__ __forceinline__ v16bf load_frag(const unsigned short* t, int ld, int row, int k0, int hi) {
  Frag f;
  const unsigned short* p = t + row * ld + k0 + hi * 8;
  f.q[0] = *reinterpret_cast<const u32x4*>(p);
  f.q[1] = *reinterpret_cast<const u32x4*>(p + 16);
  return f.v;
}

__global__ void cvt_f32_bf16(const float* __restrict__ in, unsigned short* __restrict__ out, int n) {
  int i = blockIdx.x * blockDim.x + threadIdx.x;
  int stride = gridDim.x * blockDim.x;
  for (; i < n; i += stride) out[i] = f2bf(in[i]);
}

// ---------------- QKV projection: [4096,1024] x [1024,3072]^T ----------------
// 4 waves, 64x64 tile per wave (16 WMMA per 16 b128 LDS loads per K-step: balanced)
__global__ void __launch_bounds__(128) qkv_gemm_kernel(
    const unsigned short* __restrict__ xb,   // [4096][1024] bf16
    const unsigned short* __restrict__ wb,   // [3072][1024] bf16 (row = out col)
    const float* __restrict__ bias,          // [3072]
    unsigned short* __restrict__ Qb,         // [B,H,N,D]
    unsigned short* __restrict__ Kb,         // [B,H,N,D]
    unsigned short* __restrict__ Vb) {       // [B,H,D,N]  (pre-transposed)
  __shared__ __align__(16) unsigned short At[2][128][40];
  __shared__ __align__(16) unsigned short Bt[2][128][40];
  const int mBase = blockIdx.y * 128;
  const int nBase = blockIdx.x * 128;
  const int t = threadIdx.x;
  const int lane = t & 31, l = lane & 15, hi = lane >> 4, w = t >> 5;
  const int wm = w >> 1, wn = w & 1;        // 2x2 waves, 64x64 each

  auto issue_tile = [&](int buf, int kk) {
#pragma unroll
    for (int i = 0; i < 4; ++i) {
      int c = t + i * 128;
      int row = c >> 2, col = (c & 3) * 8;
      async_copy16((unsigned)(unsigned long long)&At[buf][row][col],
                   xb + (size_t)(mBase + row) * C_ + kk + col);
      async_copy16((unsigned)(unsigned long long)&Bt[buf][row][col],
                   wb + (size_t)(nBase + row) * C_ + kk + col);
    }
  };

  v8f acc[4][4] = {};

  issue_tile(0, 0);
  wait_async_all();
  __syncthreads();

  int buf = 0;
  for (int kk = 0; kk < C_; kk += 32) {
    if (kk + 32 < C_) issue_tile(buf ^ 1, kk + 32);   // overlap DMA with compute

    const unsigned short* Ap = &At[buf][0][0];
    const unsigned short* Bp = &Bt[buf][0][0];
    v16bf a[4];
#pragma unroll
    for (int i = 0; i < 4; ++i)
      a[i] = load_frag(Ap, 40, wm * 64 + i * 16 + l, 0, hi);
#pragma unroll
    for (int j = 0; j < 4; ++j) {
      v16bf bj = load_frag(Bp, 40, wn * 64 + j * 16 + l, 0, hi);
#pragma unroll
      for (int i = 0; i < 4; ++i)
        acc[i][j] = __builtin_amdgcn_wmma_f32_16x16x32_bf16(false, a[i], false, bj, (short)0, acc[i][j], false, false);
    }
    wait_async_all();
    __syncthreads();
    buf ^= 1;
  }

#pragma unroll
  for (int i = 0; i < 4; ++i)
#pragma unroll
    for (int j = 0; j < 4; ++j)
#pragma unroll
      for (int r = 0; r < 8; ++r) {
        int m = mBase + wm * 64 + i * 16 + r + 8 * hi;   // token row
        int n = nBase + wn * 64 + j * 16 + l;            // projection col
        unsigned short bf = f2bf(acc[i][j][r] + bias[n]);
        int which = n >> 10, h = (n >> 6) & 15, d = n & 63;
        int b = m >> 11, nn = m & (N_ - 1);
        size_t qk = (((size_t)(b * H_ + h)) * N_ + nn) * D_ + d;
        if (which == 0)      Qb[qk] = bf;
        else if (which == 1) Kb[qk] = bf;
        else                 Vb[(((size_t)(b * H_ + h)) * D_ + d) * N_ + nn] = bf;
      }
}

// ---------------- Flash attention: per (b,h), 128-row Q blocks ---------------
// 4 waves x 32 rows; Q fragments cached in VGPRs for the whole kernel.
__global__ void __launch_bounds__(128) flash_attn_kernel(
    const unsigned short* __restrict__ Qb,   // [B,H,N,D]
    const unsigned short* __restrict__ Kb,   // [B,H,N,D]
    const unsigned short* __restrict__ Vb,   // [B,H,D,N]
    unsigned short* __restrict__ Ob) {       // [B,N,C] bf16
  __shared__ __align__(16) unsigned short Kt[2][64][72];
  __shared__ __align__(16) unsigned short Vt[2][64][72]; // [d][key]
  __shared__ __align__(16) unsigned short Pt[4][32][72];

  const int bh = blockIdx.y;
  const int rowBase = blockIdx.x * 128;
  const int t = threadIdx.x;
  const int lane = t & 31, l = lane & 15, hi = lane >> 4, w = t >> 5;
  const float scale = 0.125f;  // D^-0.5

  const unsigned short* Qg = Qb + ((size_t)bh * N_ + rowBase) * D_;
  const unsigned short* Kg = Kb + (size_t)bh * N_ * D_;
  const unsigned short* Vg = Vb + (size_t)bh * D_ * N_;

  auto issue_kv = [&](int buf, int ch) {
#pragma unroll
    for (int i = 0; i < 4; ++i) {
      int c = t + i * 128;
      int row = c >> 3, col = (c & 7) * 8;
      async_copy16((unsigned)(unsigned long long)&Kt[buf][row][col],
                   Kg + (size_t)(ch * 64 + row) * D_ + col);
      async_copy16((unsigned)(unsigned long long)&Vt[buf][row][col],
                   Vg + (size_t)row * N_ + ch * 64 + col);
    }
  };

  issue_kv(0, 0);

  // Q A-fragments straight from global (2x16B contiguous per lane), cached in VGPRs
  v16bf qf[2][2];
#pragma unroll
  for (int i = 0; i < 2; ++i)
#pragma unroll
    for (int ksi = 0; ksi < 2; ++ksi)
      qf[i][ksi] = load_frag(Qg, D_, w * 32 + i * 16 + l, ksi * 32, hi);

  float m_i[16], l_i[16];
  v8f Ov[2][4] = {};
#pragma unroll
  for (int r = 0; r < 16; ++r) { m_i[r] = -3.0e38f; l_i[r] = 0.f; }

  wait_async_all();
  __syncthreads();

  int buf = 0;
  for (int ch = 0; ch < N_ / 64; ++ch) {
    if (ch + 1 < N_ / 64) {
      issue_kv(buf ^ 1, ch + 1);   // overlap next chunk DMA with this chunk's math
      __builtin_prefetch(Kg + (size_t)((ch + 1) * 64) * D_, 0, 1);
    }

    // S = Q(32x64) @ K^T(64x64) per wave
    v8f sc[2][4] = {};
#pragma unroll
    for (int ksi = 0; ksi < 2; ++ksi)
#pragma unroll
      for (int c = 0; c < 4; ++c) {
        v16bf bk = load_frag(&Kt[buf][0][0], 72, c * 16 + l, ksi * 32, hi);
        sc[0][c] = __builtin_amdgcn_wmma_f32_16x16x32_bf16(false, qf[0][ksi], false, bk, (short)0, sc[0][c], false, false);
        sc[1][c] = __builtin_amdgcn_wmma_f32_16x16x32_bf16(false, qf[1][ksi], false, bk, (short)0, sc[1][c], false, false);
      }

    // online softmax; row (i*16 + r + 8*hi) lives across lanes l=0..15 of each half-wave
#pragma unroll
    for (int i = 0; i < 2; ++i)
#pragma unroll
      for (int r = 0; r < 8; ++r) {
        int s = i * 8 + r;
        float raw = fmaxf(fmaxf(sc[i][0][r], sc[i][1][r]), fmaxf(sc[i][2][r], sc[i][3][r]));
        raw = fmaxf(raw, __shfl_xor(raw, 1, 32));
        raw = fmaxf(raw, __shfl_xor(raw, 2, 32));
        raw = fmaxf(raw, __shfl_xor(raw, 4, 32));
        raw = fmaxf(raw, __shfl_xor(raw, 8, 32));
        float newm = fmaxf(m_i[s], raw * scale);
        float alpha = __expf(m_i[s] - newm);
        float psum = 0.f;
        int prow = i * 16 + r + 8 * hi;
#pragma unroll
        for (int c = 0; c < 4; ++c) {
          float p = __expf(sc[i][c][r] * scale - newm);
          psum += p;
          Pt[w][prow][c * 16 + l] = f2bf(p);
        }
        psum += __shfl_xor(psum, 1, 32);
        psum += __shfl_xor(psum, 2, 32);
        psum += __shfl_xor(psum, 4, 32);
        psum += __shfl_xor(psum, 8, 32);
        l_i[s] = alpha * l_i[s] + psum;
        m_i[s] = newm;
#pragma unroll
        for (int c2 = 0; c2 < 4; ++c2) Ov[i][c2][r] *= alpha;
      }
    __syncthreads();   // Pt write -> A-frag read ordering

    // O += P(32x64) @ V(64x64); Vt is [d][key] so B-frag loads are contiguous
#pragma unroll
    for (int ksi = 0; ksi < 2; ++ksi) {
      v16bf ap0 = load_frag(&Pt[w][0][0], 72, l,      ksi * 32, hi);
      v16bf ap1 = load_frag(&Pt[w][0][0], 72, 16 + l, ksi * 32, hi);
#pragma unroll
      for (int c2 = 0; c2 < 4; ++c2) {
        v16bf bv = load_frag(&Vt[buf][0][0], 72, c2 * 16 + l, ksi * 32, hi);
        Ov[0][c2] = __builtin_amdgcn_wmma_f32_16x16x32_bf16(false, ap0, false, bv, (short)0, Ov[0][c2], false, false);
        Ov[1][c2] = __builtin_amdgcn_wmma_f32_16x16x32_bf16(false, ap1, false, bv, (short)0, Ov[1][c2], false, false);
      }
    }

    wait_async_all();   // next chunk's DMA landed
    __syncthreads();    // and every wave is done reading the old buffers
    buf ^= 1;
  }

  const int b = bh >> 4, h = bh & 15;
#pragma unroll
  for (int i = 0; i < 2; ++i)
#pragma unroll
    for (int c2 = 0; c2 < 4; ++c2)
#pragma unroll
      for (int r = 0; r < 8; ++r) {
        int nrow = rowBase + w * 32 + i * 16 + r + 8 * hi;
        int d = c2 * 16 + l;
        float v = Ov[i][c2][r] / l_i[i * 8 + r];
        Ob[((size_t)(b * N_ + nrow)) * C_ + h * D_ + d] = f2bf(v);
      }
}

// ---------------- Output projection: [4096,1024] x [1024,1024]^T -> f32 ------
__global__ void __launch_bounds__(128) out_gemm_kernel(
    const unsigned short* __restrict__ ab,   // [4096][1024] bf16
    const unsigned short* __restrict__ wb,   // [1024][1024] bf16
    const float* __restrict__ bias,          // [1024]
    float* __restrict__ out) {               // [4096][1024] f32
  __shared__ __align__(16) unsigned short At[2][128][40];
  __shared__ __align__(16) unsigned short Bt[2][128][40];
  const int mBase = blockIdx.y * 128;
  const int nBase = blockIdx.x * 128;
  const int t = threadIdx.x;
  const int lane = t & 31, l = lane & 15, hi = lane >> 4, w = t >> 5;
  const int wm = w >> 1, wn = w & 1;

  auto issue_tile = [&](int buf, int kk) {
#pragma unroll
    for (int i = 0; i < 4; ++i) {
      int c = t + i * 128;
      int row = c >> 2, col = (c & 3) * 8;
      async_copy16((unsigned)(unsigned long long)&At[buf][row][col],
                   ab + (size_t)(mBase + row) * C_ + kk + col);
      async_copy16((unsigned)(unsigned long long)&Bt[buf][row][col],
                   wb + (size_t)(nBase + row) * C_ + kk + col);
    }
  };

  v8f acc[4][4] = {};

  issue_tile(0, 0);
  wait_async_all();
  __syncthreads();

  int buf = 0;
  for (int kk = 0; kk < C_; kk += 32) {
    if (kk + 32 < C_) issue_tile(buf ^ 1, kk + 32);

    const unsigned short* Ap = &At[buf][0][0];
    const unsigned short* Bp = &Bt[buf][0][0];
    v16bf a[4];
#pragma unroll
    for (int i = 0; i < 4; ++i)
      a[i] = load_frag(Ap, 40, wm * 64 + i * 16 + l, 0, hi);
#pragma unroll
    for (int j = 0; j < 4; ++j) {
      v16bf bj = load_frag(Bp, 40, wn * 64 + j * 16 + l, 0, hi);
#pragma unroll
      for (int i = 0; i < 4; ++i)
        acc[i][j] = __builtin_amdgcn_wmma_f32_16x16x32_bf16(false, a[i], false, bj, (short)0, acc[i][j], false, false);
    }
    wait_async_all();
    __syncthreads();
    buf ^= 1;
  }

#pragma unroll
  for (int i = 0; i < 4; ++i)
#pragma unroll
    for (int j = 0; j < 4; ++j)
#pragma unroll
      for (int r = 0; r < 8; ++r) {
        int m = mBase + wm * 64 + i * 16 + r + 8 * hi;
        int n = nBase + wn * 64 + j * 16 + l;
        out[(size_t)m * C_ + n] = acc[i][j][r] + bias[n];
      }
}

extern "C" void kernel_launch(void* const* d_in, const int* in_sizes, int n_in,
                              void* d_out, int out_size, void* d_ws, size_t ws_size,
                              hipStream_t stream) {
  (void)in_sizes; (void)n_in; (void)out_size; (void)ws_size;
  const float* x     = (const float*)d_in[0];
  const float* W_in  = (const float*)d_in[1];
  const float* b_in  = (const float*)d_in[2];
  const float* W_out = (const float*)d_in[3];
  const float* b_out = (const float*)d_in[4];
  float* out = (float*)d_out;

  char* ws = (char*)d_ws;
  unsigned short* xb  = (unsigned short*)(ws);                 //  8 MB x bf16
  unsigned short* wib = (unsigned short*)(ws + 8388608);       //  6 MB W_in bf16
  unsigned short* wob = (unsigned short*)(ws + 14680064);      //  2 MB W_out bf16
  unsigned short* Qb  = (unsigned short*)(ws + 16777216);      //  8 MB
  unsigned short* Kb  = (unsigned short*)(ws + 25165824);      //  8 MB
  unsigned short* Vb  = (unsigned short*)(ws + 33554432);      //  8 MB (transposed)
  unsigned short* Ob  = (unsigned short*)(ws + 41943040);      //  8 MB

  cvt_f32_bf16<<<1024, 256, 0, stream>>>(x, xb, B_ * N_ * C_);
  cvt_f32_bf16<<<1024, 256, 0, stream>>>(W_in, wib, 3 * C_ * C_);
  cvt_f32_bf16<<<512, 256, 0, stream>>>(W_out, wob, C_ * C_);

  qkv_gemm_kernel<<<dim3(24, 32), 128, 0, stream>>>(xb, wib, b_in, Qb, Kb, Vb);
  flash_attn_kernel<<<dim3(16, 32), 128, 0, stream>>>(Qb, Kb, Vb, Ob);
  out_gemm_kernel<<<dim3(8, 32), 128, 0, stream>>>(Ob, wob, b_out, out);
}
